// ThreeLayerGCNEncoder_19301583028532
// MI455X (gfx1250) — matrix operand order repaired and checked by reading, not compiled, not run on hardware
//
#include <hip/hip_runtime.h>
#include <hip/hip_bf16.h>

// ---------------------------------------------------------------------------
// 3-layer GCN encoder for MI455X (gfx1250, wave32, WMMA).
//   GEMMs: v_wmma_f32_16x16x32_f16 (f16 inputs, f32 accumulate),
//          4-way M register blocking: 4 WMMAs reuse each B fragment.
//   Message passing: gather + global_atomic_add_f32 scatter (L2-resident H)
//   BN: deterministic two-pass reduction
// ---------------------------------------------------------------------------

typedef __attribute__((ext_vector_type(16))) _Float16 v16h;
typedef __attribute__((ext_vector_type(8)))  _Float16 v8h;
typedef __attribute__((ext_vector_type(8)))  float    v8f;

#define GCN_N 50000
#define GCN_E 800000
#define OUTD  128
#define MTILE 64   // rows per block (4 row-tiles of 16 per wave)

// ---------------- degree / norm -------------------------------------------
__global__ void k_fill_f32(float* __restrict__ p, float v, int n) {
  int i = blockIdx.x * 256 + threadIdx.x;
  if (i < n) p[i] = v;
}

__global__ void k_deg_accum(const long long* __restrict__ dst,
                            float* __restrict__ deg, int e) {
  int i = blockIdx.x * 256 + threadIdx.x;
  if (i < e) atomicAdd(&deg[(int)dst[i]], 1.0f);
}

__global__ void k_deg_to_dinv(float* __restrict__ p, int n) {
  int i = blockIdx.x * 256 + threadIdx.x;
  if (i < n) p[i] = rsqrtf(fmaxf(p[i], 1.0f));
}

// ---------------- weight packing: W[K,128] f32 -> Wt[n][K] f16 -------------
__global__ void k_pack_w(const float* __restrict__ W, _Float16* __restrict__ Wt,
                         int K) {
  int i = blockIdx.x * 256 + threadIdx.x;   // i = k*128 + n
  if (i >= K * OUTD) return;
  int k = i >> 7;
  int n = i & 127;
  Wt[n * K + k] = (_Float16)W[i];
}

// ---------------- WMMA GEMM: C[N,128] = A[N,K] @ W[K,128] ------------------
// A: f32 row-major (lda = K, K in {512,128}), staged into LDS as f16.
// Wt: f16, n-major layout [128][K] so B fragments are contiguous 32B loads.
// Block = 256 threads = 8 wave32s. Block tile: 64 rows x 128 cols.
// Each wave owns one 16-col tile and 4 row tiles -> 4 WMMAs per B fragment.
__global__ void __launch_bounds__(256)
k_gemm_wmma(const float* __restrict__ A, const _Float16* __restrict__ Wt,
            float* __restrict__ C, int K, int N) {
  __shared__ _Float16 sA[MTILE * (512 + 8)];   // padded vs bank conflicts
  const int astride = K + 8;
  const int r0 = blockIdx.x * MTILE;
  const int t  = threadIdx.x;

  // Stage + convert the 64-row A panel (f32 -> f16) into LDS, coalesced.
  // Out-of-range rows clamp to N-1 (valid address; results discarded later).
  for (int idx = t; idx < MTILE * K; idx += 256) {
    int r = idx / K;
    int c = idx - r * K;
    int gr = r0 + r;
    if (gr >= N) gr = N - 1;
    sA[r * astride + c] = (_Float16)A[(size_t)gr * K + c];
  }
  __syncthreads();

  const int wave = t >> 5;
  const int lane = t & 31;
  const int la   = lane & 15;     // row within tile (A) / col within tile (B)
  const int hi   = lane >> 4;     // lane group
  const int aSel = hi * 16;       // A K-offset (contiguous, matches B)
  const int bSel = hi * 16;       // B K-offset (contiguous, matches A)
  const int n0   = wave * 16;     // this wave's column tile

  v8f acc[4] = {{}, {}, {}, {}};
  const _Float16* brow = Wt + (size_t)(n0 + la) * K;

  for (int kk = 0; kk < K; kk += 32) {
    // B fragment: element j -> K = kk + bSel + j (contiguous, 32B aligned);
    // loaded once, reused by 4 row tiles.
    v16h bf = *(const v16h*)(brow + kk + bSel);
#pragma unroll
    for (int rt = 0; rt < 4; ++rt) {
      const _Float16* arow = sA + (rt * 16 + la) * astride;
      // A fragment: element j -> K = kk + aSel + j (contiguous)
      v8h a0 = *(const v8h*)(arow + kk + aSel);
      v8h a1 = *(const v8h*)(arow + kk + aSel + 8);
      v16h af = __builtin_shufflevector(a0, a1,
          0, 1, 2, 3, 4, 5, 6, 7, 8, 9, 10, 11, 12, 13, 14, 15);
      acc[rt] = __builtin_amdgcn_wmma_f32_16x16x32_f16(
          /*neg_a=*/false, af, /*neg_b=*/false, bf,
          /*c_mod=*/(short)0, acc[rt], /*reuse_a=*/false, /*reuse_b=*/false);
    }
  }

  // C layout: VGPR r -> row = rt*16 + r + 8*hi, col = n0 + la
#pragma unroll
  for (int rt = 0; rt < 4; ++rt) {
    int rowBase = r0 + rt * 16 + hi * 8;
#pragma unroll
    for (int r = 0; r < 8; ++r) {
      int row = rowBase + r;
      if (row < N) C[(size_t)row * OUTD + n0 + la] = acc[rt][r];
    }
  }
}

// ---------------- zero buffer ---------------------------------------------
__global__ void k_zero(float* __restrict__ p, int n) {
  int i = blockIdx.x * 256 + threadIdx.x;
  if (i < n) p[i] = 0.0f;
}

// ---------------- edge scatter: out[dst] += H[src] * dinv[s]*dinv[d] -------
// One wave32 per edge; each lane handles 4 consecutive features (float4).
__global__ void k_scatter(const float* __restrict__ H,
                          const long long* __restrict__ src,
                          const long long* __restrict__ dst,
                          const float* __restrict__ dinv,
                          float* __restrict__ out, int e) {
  int t = blockIdx.x * 256 + threadIdx.x;
  int edge = t >> 5;
  if (edge >= e) return;
  int lane = t & 31;
  int s = (int)src[edge];
  int d = (int)dst[edge];
  float w = dinv[s] * dinv[d];
  float4 v = *(const float4*)(H + (size_t)s * OUTD + lane * 4);
  float* o = out + (size_t)d * OUTD + lane * 4;
  atomicAdd(o + 0, v.x * w);
  atomicAdd(o + 1, v.y * w);
  atomicAdd(o + 2, v.z * w);
  atomicAdd(o + 3, v.w * w);
}

// ---------------- self loop + bias: out += H * dinv^2 + b ------------------
__global__ void k_self_bias(const float* __restrict__ H,
                            const float* __restrict__ dinv,
                            const float* __restrict__ b,
                            float* __restrict__ out, int total) {
  int i = blockIdx.x * 256 + threadIdx.x;
  if (i >= total) return;
  int row = i >> 7;
  int f   = i & 127;
  float dv = dinv[row];
  out[i] += H[i] * dv * dv + b[f];
}

// ---------------- BN stats: deterministic 2-pass reduction -----------------
__global__ void k_bn_partial(const float* __restrict__ X,
                             float* __restrict__ part, int n) {
  int t = threadIdx.x;          // 256 threads
  int f = t & 127;
  int half = t >> 7;
  float s = 0.0f, q = 0.0f;
  for (int row = blockIdx.x * 2 + half; row < n; row += gridDim.x * 2) {
    float v = X[(size_t)row * OUTD + f];
    s += v;
    q += v * v;
  }
  __shared__ float sh_s[256], sh_q[256];
  sh_s[t] = s;
  sh_q[t] = q;
  __syncthreads();
  if (t < 128) {
    part[blockIdx.x * 256 + t]       = sh_s[t] + sh_s[t + 128];
    part[blockIdx.x * 256 + 128 + t] = sh_q[t] + sh_q[t + 128];
  }
}

__global__ void k_bn_final(const float* __restrict__ part,
                           float* __restrict__ stats, int n, int nblk) {
  int f = threadIdx.x;
  if (f >= 128) return;
  float s = 0.0f, q = 0.0f;
  for (int b = 0; b < nblk; ++b) {
    s += part[b * 256 + f];
    q += part[b * 256 + 128 + f];
  }
  float mean = s / (float)n;
  float var  = q / (float)n - mean * mean;
  stats[f]        = mean;
  stats[128 + f]  = rsqrtf(var + 1e-5f);
}

__global__ void k_bn_prelu(float* __restrict__ X,
                           const float* __restrict__ stats,
                           const float* __restrict__ g,
                           const float* __restrict__ be,
                           const float* __restrict__ a, int total) {
  int i = blockIdx.x * 256 + threadIdx.x;
  if (i >= total) return;
  int f = i & 127;
  float v = (X[i] - stats[f]) * stats[128 + f] * g[f] + be[f];
  X[i] = (v >= 0.0f) ? v : a[0] * v;
}

// ---------------------------------------------------------------------------
extern "C" void kernel_launch(void* const* d_in, const int* in_sizes, int n_in,
                              void* d_out, int out_size, void* d_ws,
                              size_t ws_size, hipStream_t stream) {
  const float*     x    = (const float*)d_in[0];
  const long long* ei   = (const long long*)d_in[1];   // int64 [2, E]
  const float*     W1   = (const float*)d_in[2];
  const float*     b1   = (const float*)d_in[3];
  const float*     W2   = (const float*)d_in[4];
  const float*     b2   = (const float*)d_in[5];
  const float*     W3   = (const float*)d_in[6];
  const float*     b3   = (const float*)d_in[7];
  const float*     g1   = (const float*)d_in[8];
  const float*     be1  = (const float*)d_in[9];
  const float*     g2   = (const float*)d_in[10];
  const float*     be2  = (const float*)d_in[11];
  const float*     a1   = (const float*)d_in[12];
  const float*     a2   = (const float*)d_in[13];
  float*           out  = (float*)d_out;
  char*            ws   = (char*)d_ws;

  const int N = GCN_N, E = GCN_E;
  const long long* srcp = ei;
  const long long* dstp = ei + E;

  // workspace layout (256B aligned offsets)
  float*     dinv  = (float*)(ws + 0);           // 50000 f32
  float*     Ha    = (float*)(ws + 200192);      // N*128 f32 (25.6 MB)
  float*     Hb    = (float*)(ws + 25800192);    // N*128 f32 (25.6 MB)
  _Float16*  Wt1   = (_Float16*)(ws + 51400192); // 128*512 f16
  _Float16*  Wt2   = (_Float16*)(ws + 51531264); // 128*128 f16
  _Float16*  Wt3   = (_Float16*)(ws + 51564032); // 128*128 f16
  float*     part  = (float*)(ws + 51596800);    // 256*256 f32
  float*     stats = (float*)(ws + 51858944);    // 256 f32

  const int total    = N * OUTD;                 // 6,400,000
  const int gElem    = (total + 255) / 256;      // 25000
  const int gNode    = (N + 255) / 256;          // 196
  const int gEdge    = (E + 255) / 256;          // 3125
  const int gScatter = (E * 32 + 255) / 256;     // 100000
  const int gGemm    = (N + MTILE - 1) / MTILE;  // 782

  // ---- normalization: deg (with self loop) -> dinv ----
  k_fill_f32<<<gNode, 256, 0, stream>>>(dinv, 1.0f, N);
  k_deg_accum<<<gEdge, 256, 0, stream>>>(dstp, dinv, E);
  k_deg_to_dinv<<<gNode, 256, 0, stream>>>(dinv, N);

  // ---- pack weights to f16, n-major ----
  k_pack_w<<<(512 * OUTD) / 256, 256, 0, stream>>>(W1, Wt1, 512);
  k_pack_w<<<(128 * OUTD) / 256, 256, 0, stream>>>(W2, Wt2, 128);
  k_pack_w<<<(128 * OUTD) / 256, 256, 0, stream>>>(W3, Wt3, 128);

  // ---- layer 1 ----
  k_gemm_wmma<<<gGemm, 256, 0, stream>>>(x, Wt1, Ha, 512, N);
  k_zero<<<gElem, 256, 0, stream>>>(Hb, total);
  k_scatter<<<gScatter, 256, 0, stream>>>(Ha, srcp, dstp, dinv, Hb, E);
  k_self_bias<<<gElem, 256, 0, stream>>>(Ha, dinv, b1, Hb, total);
  k_bn_partial<<<256, 256, 0, stream>>>(Hb, part, N);
  k_bn_final<<<1, 128, 0, stream>>>(part, stats, N, 256);
  k_bn_prelu<<<gElem, 256, 0, stream>>>(Hb, stats, g1, be1, a1, total);

  // ---- layer 2 ----
  k_gemm_wmma<<<gGemm, 256, 0, stream>>>(Hb, Wt2, Ha, 128, N);
  k_zero<<<gElem, 256, 0, stream>>>(Hb, total);
  k_scatter<<<gScatter, 256, 0, stream>>>(Ha, srcp, dstp, dinv, Hb, E);
  k_self_bias<<<gElem, 256, 0, stream>>>(Ha, dinv, b2, Hb, total);
  k_bn_partial<<<256, 256, 0, stream>>>(Hb, part, N);
  k_bn_final<<<1, 128, 0, stream>>>(part, stats, N, 256);
  k_bn_prelu<<<gElem, 256, 0, stream>>>(Hb, stats, g2, be2, a2, total);

  // ---- layer 3 (no BN/PReLU) ----
  k_gemm_wmma<<<gGemm, 256, 0, stream>>>(Hb, Wt3, Ha, 128, N);
  k_zero<<<gElem, 256, 0, stream>>>(out, total);
  k_scatter<<<gScatter, 256, 0, stream>>>(Ha, srcp, dstp, dinv, out, E);
  k_self_bias<<<gElem, 256, 0, stream>>>(Ha, dinv, b3, out, total);
}
